// Attention_24343874633732
// MI455X (gfx1250) — compile-verified, matrix operand order checked by reading, not run on hardware
//
#include <hip/hip_runtime.h>

// Problem constants (fixed by the reference)
#define B_  4
#define S_  1024
#define D_  2048
#define H_  32
#define KV_ 8
#define HD_ 64

typedef __bf16 bf16;
typedef __attribute__((ext_vector_type(8)))  bf16  v8bf;
typedef __attribute__((ext_vector_type(16))) bf16  v16bf;
typedef __attribute__((ext_vector_type(8)))  float v8f;
typedef __attribute__((ext_vector_type(4)))  float v4f;

// Load one wave32 WMMA A/B fragment (16x32 bf16, contraction-major runs of 8):
// per lane: 8 contiguous bf16 at p and 8 at p+16 (two 16B loads).
static __device__ inline v16bf ld_frag(const bf16* p) {
    v8bf lo = *(const v8bf*)p;
    v8bf hi = *(const v8bf*)(p + 16);
    v16bf r;
#pragma unroll
    for (int i = 0; i < 8; ++i) { r[i] = lo[i]; r[i + 8] = hi[i]; }
    return r;
}

static __device__ inline v8f wmma_bf16(v16bf a, v16bf b, v8f c) {
    // (neg_a, A, neg_b, B, c_mod, C, reuse_a, reuse_b)
    return __builtin_amdgcn_wmma_f32_16x16x32_bf16(false, a, false, b, (short)0, c,
                                                   false, false);
}

// ---------------------------------------------------------------- cast f32->bf16
// 8 elements/thread: two b128 loads -> one b128 store (full line-rate conversion).
__global__ void k_cast8(const float* __restrict__ in, bf16* __restrict__ out, int n8) {
    int i = blockIdx.x * blockDim.x + threadIdx.x;
    if (i >= n8) return;
    size_t base = (size_t)i * 8;
    v4f a = *(const v4f*)(in + base);
    v4f b = *(const v4f*)(in + base + 4);
    v8bf r;
#pragma unroll
    for (int j = 0; j < 4; ++j) { r[j] = (bf16)a[j]; r[j + 4] = (bf16)b[j]; }
    *(v8bf*)(out + base) = r;
}

// ---------------------------------------------------------------- generic GEMM
// C[M,N] (f32, row-major) = A[M,K] (bf16 row-major) * W[N,K]^T (bf16 row-major)
// 128 threads = 4 waves; block tile 64x128, wave tile 32x64 (2x4 WMMA frags):
// 8 v_wmma per 12 b128 loads per k-step; k-loop unrolled 2x so the scheduler can
// double-buffer fragments and overlap next-iteration loads with current WMMAs.
__global__ __launch_bounds__(128) void k_gemm_bf16(const bf16* __restrict__ A,
                                                   const bf16* __restrict__ W,
                                                   float* __restrict__ C,
                                                   int M, int N, int K) {
    const int lane  = threadIdx.x & 31;
    const int wave  = threadIdx.x >> 5;
    const int n16   = lane & 15;
    const int half  = lane >> 4;
    const int kbase = half * 8;
    const int bm = blockIdx.y * 64  + (wave >> 1) * 32;
    const int bn = blockIdx.x * 128 + (wave & 1) * 64;

    v8f acc[2][4];
#pragma unroll
    for (int i = 0; i < 2; ++i)
#pragma unroll
        for (int j = 0; j < 4; ++j)
#pragma unroll
            for (int r = 0; r < 8; ++r) acc[i][j][r] = 0.0f;

    const size_t rowA0 = (size_t)(bm + n16) * K;
    const size_t rowA1 = (size_t)(bm + 16 + n16) * K;
    size_t rowW[4];
#pragma unroll
    for (int j = 0; j < 4; ++j) rowW[j] = (size_t)(bn + j * 16 + n16) * K;

#pragma unroll 2
    for (int k0 = 0; k0 < K; k0 += 32) {
        v16bf a0 = ld_frag(A + rowA0 + k0 + kbase);
        v16bf a1 = ld_frag(A + rowA1 + k0 + kbase);
        v16bf bfr[4];
#pragma unroll
        for (int j = 0; j < 4; ++j) bfr[j] = ld_frag(W + rowW[j] + k0 + kbase);
#pragma unroll
        for (int j = 0; j < 4; ++j) {
            acc[0][j] = wmma_bf16(a0, bfr[j], acc[0][j]);
            acc[1][j] = wmma_bf16(a1, bfr[j], acc[1][j]);
        }
    }

#pragma unroll
    for (int i = 0; i < 2; ++i)
#pragma unroll
        for (int j = 0; j < 4; ++j)
#pragma unroll
            for (int r = 0; r < 8; ++r) {
                int row = bm + i * 16 + r + half * 8;
                int col = bn + j * 16 + n16;
                C[(size_t)row * N + col] = acc[i][j][r];
            }
}

// ---------------------------------------------------------------- RoPE (head-indexed!)
// Reference quirk: rotation angle row = HEAD index (cos[:nh]), same for all positions.
// Q: (B,S,H,HD) f32 -> (B,H,S,HD) bf16
__global__ void k_rope_q(const float* __restrict__ qf, const float* __restrict__ ct,
                         const float* __restrict__ st, bf16* __restrict__ qb) {
    int idx = blockIdx.x * blockDim.x + threadIdx.x;      // B*S*H*(HD/2) = 2^22
    int j = idx & 31;
    int h = (idx >> 5) & 31;
    int s = (idx >> 10) & 1023;
    int b = idx >> 20;
    size_t src = ((size_t)((b * S_ + s) * H_ + h)) * HD_ + 2 * j;
    float t0 = qf[src], t1 = qf[src + 1];
    float c = ct[h * (HD_ / 2) + j], sn = st[h * (HD_ / 2) + j];
    size_t dst = ((size_t)((b * H_ + h) * S_ + s)) * HD_ + 2 * j;
    qb[dst]     = (bf16)(t0 * c - t1 * sn);
    qb[dst + 1] = (bf16)(t0 * sn + t1 * c);
}

// K: (B,S,KV,HD) f32 -> (B,KV,S,HD) bf16
__global__ void k_rope_k(const float* __restrict__ kf, const float* __restrict__ ct,
                         const float* __restrict__ st, bf16* __restrict__ kb) {
    int idx = blockIdx.x * blockDim.x + threadIdx.x;      // B*S*KV*(HD/2) = 2^20
    int j  = idx & 31;
    int kv = (idx >> 5) & 7;
    int s  = (idx >> 8) & 1023;
    int b  = idx >> 18;
    size_t src = ((size_t)((b * S_ + s) * KV_ + kv)) * HD_ + 2 * j;
    float t0 = kf[src], t1 = kf[src + 1];
    float c = ct[kv * (HD_ / 2) + j], sn = st[kv * (HD_ / 2) + j];
    size_t dst = ((size_t)((b * KV_ + kv) * S_ + s)) * HD_ + 2 * j;
    kb[dst]     = (bf16)(t0 * c - t1 * sn);
    kb[dst + 1] = (bf16)(t0 * sn + t1 * c);
}

// V: (B,S,KV,HD) f32 -> (B,KV,HD,S) bf16  (transposed so PV B-operand is contiguous)
__global__ void k_vtrans(const float* __restrict__ vf, bf16* __restrict__ vt) {
    int idx = blockIdx.x * blockDim.x + threadIdx.x;      // B*S*KV*HD = 2^21
    int d  = idx & 63;
    int kv = (idx >> 6) & 7;
    int s  = (idx >> 9) & 1023;
    int b  = idx >> 19;
    size_t src = ((size_t)((b * S_ + s) * KV_ + kv)) * HD_ + d;
    size_t dst = ((size_t)((b * KV_ + kv) * HD_ + d)) * S_ + s;
    vt[dst] = (bf16)vf[src];
}

// ---------------------------------------------------------------- flash attention
// One wave per (b, h, 16-query tile). Online softmax in f32, WMMA bf16 for S=QK^T
// and O=PV. Causal mask applied analytically (reference 'mask' input unused).
__global__ __launch_bounds__(32) void k_attn(const bf16* __restrict__ Q,
                                             const bf16* __restrict__ Kb,
                                             const bf16* __restrict__ Vt,
                                             bf16* __restrict__ O) {
    __shared__ __align__(16) bf16 pshm[16 * 32];   // P tile staging (C-layout -> A-layout)
    const int qt = blockIdx.x, h = blockIdx.y, b = blockIdx.z;
    const int kvh = h >> 2;                        // GQA: rep = H/KV = 4
    const int lane = threadIdx.x;
    const int n16 = lane & 15, half = lane >> 4, kbase = half * 8;
    const int q0 = qt * 16;

    // Q fragments: 16 rows x 64 (two 16x32 A tiles over head_dim)
    const bf16* qrow = Q + ((size_t)((b * H_ + h) * S_ + q0 + n16)) * HD_;
    v16bf qa0 = ld_frag(qrow + kbase);
    v16bf qa1 = ld_frag(qrow + 32 + kbase);

    v8f acc[4];
    float mrun[8], lrun[8];
#pragma unroll
    for (int c = 0; c < 4; ++c)
#pragma unroll
        for (int r = 0; r < 8; ++r) acc[c][r] = 0.0f;
#pragma unroll
    for (int r = 0; r < 8; ++r) { mrun[r] = -1e30f; lrun[r] = 0.0f; }

    for (int kb0 = 0; kb0 < q0 + 16; kb0 += 32) {     // causal: keys <= q0+15
        // K fragments for two 16-key subtiles x two d-chunks
        const bf16* krow = Kb + ((size_t)((b * KV_ + kvh) * S_ + kb0 + n16)) * HD_;
        v16bf k00 = ld_frag(krow + kbase);
        v16bf k01 = ld_frag(krow + 32 + kbase);
        v16bf k10 = ld_frag(krow + 16 * HD_ + kbase);
        v16bf k11 = ld_frag(krow + 16 * HD_ + 32 + kbase);

        v8f s0, s1;
#pragma unroll
        for (int r = 0; r < 8; ++r) { s0[r] = 0.0f; s1[r] = 0.0f; }
        s0 = wmma_bf16(qa0, k00, s0);
        s0 = wmma_bf16(qa1, k01, s0);
        s1 = wmma_bf16(qa0, k10, s1);
        s1 = wmma_bf16(qa1, k11, s1);

#pragma unroll
        for (int r = 0; r < 8; ++r) {
            int grow = q0 + r + half * 8;
            float v0 = s0[r] * 0.125f;               // 1/sqrt(64)
            float v1 = s1[r] * 0.125f;
            if (kb0 + n16 > grow)       v0 = -1e30f; // causal mask
            if (kb0 + 16 + n16 > grow)  v1 = -1e30f;
            // row max across 16 columns (xor 1/2/4/8 stays within each lane-half)
            float rm = fmaxf(v0, v1);
            rm = fmaxf(rm, __shfl_xor(rm, 1, 32));
            rm = fmaxf(rm, __shfl_xor(rm, 2, 32));
            rm = fmaxf(rm, __shfl_xor(rm, 4, 32));
            rm = fmaxf(rm, __shfl_xor(rm, 8, 32));
            float mnew  = fmaxf(mrun[r], rm);
            float alpha = __expf(mrun[r] - mnew);
            mrun[r] = mnew;
            float p0 = __expf(v0 - mnew);
            float p1 = __expf(v1 - mnew);
            float rs = p0 + p1;
            rs += __shfl_xor(rs, 1, 32);
            rs += __shfl_xor(rs, 2, 32);
            rs += __shfl_xor(rs, 4, 32);
            rs += __shfl_xor(rs, 8, 32);
            lrun[r] = lrun[r] * alpha + rs;
#pragma unroll
            for (int c = 0; c < 4; ++c) acc[c][r] *= alpha;
            // stage P (C-layout element (r+half*8, sub*16+n16)) as bf16 in LDS
            pshm[(r + half * 8) * 32 + n16]      = (bf16)p0;
            pshm[(r + half * 8) * 32 + 16 + n16] = (bf16)p1;
        }
        __syncthreads();                              // single-wave WG: cheap DS fence
        v16bf pa = ld_frag(&pshm[n16 * 32 + kbase]);  // re-read P in A-layout

        // O += P * V : B-operand contiguous over keys thanks to (B,KV,HD,S) layout
        const bf16* vbase = Vt + ((size_t)((b * KV_ + kvh) * HD_)) * S_ + kb0 + kbase;
#pragma unroll
        for (int c = 0; c < 4; ++c) {
            v16bf vb = ld_frag(vbase + (size_t)(c * 16 + n16) * S_);
            acc[c] = wmma_bf16(pa, vb, acc[c]);
        }
        __syncthreads();                              // protect pshm for next iter
    }

    // epilogue: normalize and store to (B,S,H*HD) bf16 for the output projection
#pragma unroll
    for (int r = 0; r < 8; ++r) {
        float inv = 1.0f / lrun[r];
        int row = q0 + r + half * 8;
        bf16* orow = O + ((size_t)(b * S_ + row)) * (H_ * HD_) + h * HD_;
#pragma unroll
        for (int c = 0; c < 4; ++c) orow[c * 16 + n16] = (bf16)(acc[c][r] * inv);
    }
}

// ---------------------------------------------------------------- host launcher
extern "C" void kernel_launch(void* const* d_in, const int* in_sizes, int n_in,
                              void* d_out, int out_size, void* d_ws, size_t ws_size,
                              hipStream_t stream) {
    (void)in_sizes; (void)n_in; (void)out_size; (void)ws_size;
    const float* x  = (const float*)d_in[0];
    const float* fc = (const float*)d_in[1];
    const float* fs = (const float*)d_in[2];
    /* d_in[3] = additive mask, unused (causal handled analytically) */
    const float* wq = (const float*)d_in[4];
    const float* wk = (const float*)d_in[5];
    const float* wv = (const float*)d_in[6];
    const float* wo = (const float*)d_in[7];

    const int M  = B_ * S_;           // 4096
    char* ws = (char*)d_ws;
    size_t off = 0;
    auto take = [&](size_t bytes) -> char* {
        char* p = ws + off;
        off += (bytes + 255) & ~(size_t)255;
        return p;
    };
    bf16*  xb    = (bf16*) take((size_t)M * D_ * 2);            // x bf16
    bf16*  wqb   = (bf16*) take((size_t)(H_*HD_)  * D_ * 2);    // wq bf16
    bf16*  wkb   = (bf16*) take((size_t)(KV_*HD_) * D_ * 2);    // wk bf16
    bf16*  wvb   = (bf16*) take((size_t)(KV_*HD_) * D_ * 2);    // wv bf16
    bf16*  wob   = (bf16*) take((size_t)D_ * D_ * 2);           // wo bf16
    float* qf    = (float*)take((size_t)M * (H_*HD_)  * 4);     // q  f32 (B,S,H,HD)
    float* kf    = (float*)take((size_t)M * (KV_*HD_) * 4);     // k  f32 (B,S,KV,HD)
    float* vf    = (float*)take((size_t)M * (KV_*HD_) * 4);     // v  f32 (B,S,KV,HD)
    bf16*  qbb   = (bf16*) take((size_t)M * (H_*HD_)  * 2);     // q  bf16 (B,H,S,HD)
    bf16*  kbb   = (bf16*) take((size_t)M * (KV_*HD_) * 2);     // k  bf16 (B,KV,S,HD)
    bf16*  vtb   = (bf16*) take((size_t)M * (KV_*HD_) * 2);     // v  bf16 (B,KV,HD,S)
    bf16*  attnb = (bf16*) take((size_t)M * (H_*HD_)  * 2);     // attn out bf16 (B,S,H*HD)

    // 1) casts to bf16 (8 elements per thread; all sizes divisible by 8)
    {
        int n8;
        n8 = (M * D_) / 8;
        k_cast8<<<(n8 + 255) / 256, 256, 0, stream>>>(x,  xb,  n8);
        n8 = ((H_*HD_)  * D_) / 8;
        k_cast8<<<(n8 + 255) / 256, 256, 0, stream>>>(wq, wqb, n8);
        n8 = ((KV_*HD_) * D_) / 8;
        k_cast8<<<(n8 + 255) / 256, 256, 0, stream>>>(wk, wkb, n8);
        n8 = ((KV_*HD_) * D_) / 8;
        k_cast8<<<(n8 + 255) / 256, 256, 0, stream>>>(wv, wvb, n8);
        n8 = (D_ * D_) / 8;
        k_cast8<<<(n8 + 255) / 256, 256, 0, stream>>>(wo, wob, n8);
    }
    // 2) QKV projections (WMMA GEMM, block tile 64x128)
    k_gemm_bf16<<<dim3((H_*HD_)/128,  M/64), 128, 0, stream>>>(xb, wqb, qf, M, H_*HD_,  D_);
    k_gemm_bf16<<<dim3((KV_*HD_)/128, M/64), 128, 0, stream>>>(xb, wkb, kf, M, KV_*HD_, D_);
    k_gemm_bf16<<<dim3((KV_*HD_)/128, M/64), 128, 0, stream>>>(xb, wvb, vf, M, KV_*HD_, D_);
    // 3) RoPE (+layout transforms to bf16)
    k_rope_q<<<(B_*S_*H_ *(HD_/2)) / 256, 256, 0, stream>>>(qf, fc, fs, qbb);
    k_rope_k<<<(B_*S_*KV_*(HD_/2)) / 256, 256, 0, stream>>>(kf, fc, fs, kbb);
    k_vtrans<<<(B_*S_*KV_* HD_)    / 256, 256, 0, stream>>>(vf, vtb);
    // 4) causal flash attention (WMMA scores + PV)
    k_attn<<<dim3(S_/16, H_, B_), 32, 0, stream>>>(qbb, kbb, vtb, attnb);
    // 5) output projection into d_out (f32)
    k_gemm_bf16<<<dim3(D_/128, M/64), 128, 0, stream>>>(attnb, wob, (float*)d_out,
                                                        M, D_, D_);
}
